// Seq2SeqDecoder_88802743812805
// MI455X (gfx1250) — compile-verified, matrix operand order checked by reading, not run on hardware
//
#include <hip/hip_runtime.h>
#include <math.h>

// ---------------- types for WMMA ----------------
typedef float v8f  __attribute__((ext_vector_type(8)));
typedef __bf16 v16bf __attribute__((ext_vector_type(16)));
typedef unsigned int v4u __attribute__((ext_vector_type(4)));

union Frag {           // 32 bytes: one 16x32 bf16 A/B fragment (8 VGPRs)
    v16bf v;
    v4u   q[2];
};

__device__ inline unsigned short f32_to_bf16(float f) {
    unsigned int u = __float_as_uint(f);
    u += 0x7FFFu + ((u >> 16) & 1u);     // round-to-nearest-even
    return (unsigned short)(u >> 16);
}

__device__ inline float sigm(float x) { return 1.0f / (1.0f + __expf(-x)); }

// CDNA5 async global->LDS copy (16B per lane), tracked by ASYNCcnt.
__device__ inline void async_copy16(unsigned lds_off, const void* gaddr) {
    asm volatile("global_load_async_to_lds_b128 %0, %1, off"
                 :: "v"(lds_off), "v"(gaddr)
                 : "memory");
}
__device__ inline void wait_async0() {
    asm volatile("s_wait_asynccnt 0x0" ::: "memory");
}

// ---------------- setup kernels ----------------
__global__ void f32_to_bf16_kernel(const float* __restrict__ src,
                                   unsigned short* __restrict__ dst, int n) {
    for (int i = blockIdx.x * blockDim.x + threadIdx.x; i < n;
         i += gridDim.x * blockDim.x)
        dst[i] = f32_to_bf16(src[i]);
}

// h0 = tile(encoder_finals, L) ; c0 = 0. hbuf layout: [layer][pingpong][b*1024+j]
__global__ void init_state_kernel(const float* __restrict__ enc,
                                  unsigned short* __restrict__ hbuf,
                                  float* __restrict__ hT,
                                  float* __restrict__ cT) {
    int idx = blockIdx.x * blockDim.x + threadIdx.x;   // over 3*64*1024
    if (idx >= 3 * 65536) return;
    int l  = idx >> 16;
    int bj = idx & 65535;
    float v = enc[bj];
    unsigned short hb = f32_to_bf16(v);
    hbuf[(l * 2 + 0) * 65536 + bj] = hb;
    hbuf[(l * 2 + 1) * 65536 + bj] = hb;
    hT[idx] = v;
    cT[idx] = 0.0f;
}

// ---------------- fused LSTM layer-step ----------------
// Grid: 16 blocks (each owns 64 h-columns), 256 threads (8 wave32).
// gates(64 x 256local) = A_x(64x1024)@Wih^T + h_prev(64x1024)@Whh^T, then cell.
__global__ void __launch_bounds__(256)
lstm_step_kernel(const unsigned short* __restrict__ Ax, long long ax_stride,
                 const unsigned short* __restrict__ Wih,
                 const unsigned short* __restrict__ Whh,
                 const float* __restrict__ bias,
                 const unsigned short* __restrict__ hprev,
                 float* __restrict__ cstate,
                 unsigned short* __restrict__ hout_bf,
                 float* __restrict__ hT,
                 float* __restrict__ out2, long long out2_stride) {
    __shared__ float s_gates[64 * 256];                          // 64 KB
    __shared__ __align__(16) unsigned short s_a[2][64 * 32];     // 2 x 4 KB A slabs

    const int tid  = threadIdx.x;
    const int wid  = tid >> 5;          // wave id 0..7
    const int lane = tid & 31;
    const int ln   = lane & 15;
    const int sel  = (lane < 16) ? 0 : 8;   // K-half select (A/B 16-bit layout)
    const int jb   = blockIdx.x;        // 0..15 : owns h columns [jb*64, jb*64+64)

    // per-thread staging slot: 16B = 8 halfs
    const int srow = tid >> 2;                     // 0..63
    const int sseg = (tid & 3) * 8;                // 0,8,16,24
    const unsigned lds_dst[2] = {
        (unsigned)(size_t)&s_a[0][srow * 32 + sseg],
        (unsigned)(size_t)&s_a[1][srow * 32 + sseg]
    };

    v8f acc[4][2];
#pragma unroll
    for (int mt = 0; mt < 4; ++mt)
#pragma unroll
        for (int j = 0; j < 2; ++j) acc[mt][j] = (v8f)0.0f;

    // phase 0: x @ Wih^T     phase 1: h_prev @ Whh^T
    for (int phase = 0; phase < 2; ++phase) {
        const unsigned short* Aptr = phase ? hprev : Ax;
        const long long astride    = phase ? 1024 : ax_stride;
        const unsigned short* Wptr = phase ? Whh : Wih;

        const unsigned short* sgp = Aptr + (long long)srow * astride + sseg;

        // per-wave weight row pointers for the 2 N-tiles (B = W^T)
        const unsigned short* wrB[2];
#pragma unroll
        for (int j = 0; j < 2; ++j) {
            int nt   = wid * 2 + j;     // 0..15 ; quadrant q = nt>>2
            int wrow = (nt >> 2) * 1024 + jb * 64 + (nt & 3) * 16 + ln;
            wrB[j] = Wptr + (long long)wrow * 1024 + sel;
        }

        // prologue: stage K-step 0 into slab 0; preload its B fragments
        async_copy16(lds_dst[0], sgp);
        Frag bfr[2][2];
#pragma unroll
        for (int j = 0; j < 2; ++j) {
            bfr[0][j].q[0] = *(const v4u*)(wrB[j]);
            bfr[0][j].q[1] = *(const v4u*)(wrB[j] + 16);
        }

#pragma unroll
        for (int k = 0; k < 32; ++k) {
            const int cur = k & 1;
            const int nxt = cur ^ 1;
            // own stage-k complete -> barrier makes it visible to all waves,
            // and guarantees everyone finished reading slab[nxt] (iter k-1).
            wait_async0();
            __syncthreads();
            if (k + 1 < 32)
                async_copy16(lds_dst[nxt], sgp + (k + 1) * 32);

            // prefetch next B fragments (overlaps with WMMA below)
            if (k + 1 < 32) {
#pragma unroll
                for (int j = 0; j < 2; ++j) {
                    bfr[nxt][j].q[0] = *(const v4u*)(wrB[j] + (k + 1) * 32);
                    bfr[nxt][j].q[1] = *(const v4u*)(wrB[j] + (k + 1) * 32 + 16);
                }
            }

#pragma unroll
            for (int mt = 0; mt < 4; ++mt) {
                Frag afrag;
                const unsigned short* ar = &s_a[cur][(mt * 16 + ln) * 32];
                afrag.q[0] = *(const v4u*)(ar + sel);
                afrag.q[1] = *(const v4u*)(ar + 16 + sel);
#pragma unroll
                for (int j = 0; j < 2; ++j)
                    acc[mt][j] = __builtin_amdgcn_wmma_f32_16x16x32_bf16(
                        false, afrag.v, false, bfr[cur][j].v,
                        (short)0, acc[mt][j], false, false);
            }
        }
    }

    __syncthreads();
    // spill accumulators: C layout VGPR r -> M = r (lanes<16) / 8+r (lanes>=16)
#pragma unroll
    for (int mt = 0; mt < 4; ++mt)
#pragma unroll
        for (int j = 0; j < 2; ++j) {
            int nt   = wid * 2 + j;
            int colb = nt * 16 + ln;        // local col = q*64 + c
            int rowb = mt * 16 + sel;
#pragma unroll
            for (int r = 0; r < 8; ++r)
                s_gates[(rowb + r) * 256 + colb] = acc[mt][j][r];
        }
    __syncthreads();

    // fused LSTM cell for our 64 columns
    for (int idx = tid; idx < 64 * 64; idx += 256) {
        int row = idx >> 6;                 // batch
        int c   = idx & 63;
        int j   = jb * 64 + c;              // h column
        float ig = s_gates[row * 256 +       c] + bias[         j];
        float fg = s_gates[row * 256 +  64 + c] + bias[1024 +   j];
        float gg = s_gates[row * 256 + 128 + c] + bias[2048 +   j];
        float og = s_gates[row * 256 + 192 + c] + bias[3072 +   j];
        float cp = cstate[row * 1024 + j];
        float cn = sigm(fg) * cp + sigm(ig) * tanhf(gg);
        float hn = sigm(og) * tanhf(cn);
        cstate[row * 1024 + j]  = cn;
        hout_bf[row * 1024 + j] = f32_to_bf16(hn);
        hT[row * 1024 + j]      = hn;
        if (out2) out2[(long long)row * out2_stride + j] = hn;
    }
}

// ---------------- host ----------------
extern "C" void kernel_launch(void* const* d_in, const int* in_sizes, int n_in,
                              void* d_out, int out_size, void* d_ws, size_t ws_size,
                              hipStream_t stream) {
    const float* inputs = (const float*)d_in[0];   // (64,256,1024)
    const float* enc    = (const float*)d_in[1];   // (1,64,1024)
    const float* W_ih   = (const float*)d_in[2];   // (3,4096,1024)
    const float* W_hh   = (const float*)d_in[3];   // (3,4096,1024)
    const float* bias   = (const float*)d_in[4];   // (3,4096)

    float* out     = (float*)d_out;
    float* hT_base = out;                      // (3,64,1024)
    float* cT_base = out + 3 * 65536;          // (3,64,1024)  (live c state)
    float* pre_out = out + 6 * 65536;          // (64,256,1024)

    char* ws = (char*)d_ws;                    // needs ~81 MB
    unsigned short* Xb   = (unsigned short*)(ws);               // 32 MB
    unsigned short* Wihb = (unsigned short*)(ws + 33554432);    // 24 MB
    unsigned short* Whhb = (unsigned short*)(ws + 58720256);    // 24 MB
    unsigned short* hbuf = (unsigned short*)(ws + 83886080);    // 768 KB

    f32_to_bf16_kernel<<<2048, 256, 0, stream>>>(inputs, Xb,   64 * 256 * 1024);
    f32_to_bf16_kernel<<<2048, 256, 0, stream>>>(W_ih,   Wihb, 3 * 4096 * 1024);
    f32_to_bf16_kernel<<<2048, 256, 0, stream>>>(W_hh,   Whhb, 3 * 4096 * 1024);
    init_state_kernel<<<(3 * 65536 + 255) / 256, 256, 0, stream>>>(
        enc, hbuf, hT_base, cT_base);

    for (int t = 0; t < 256; ++t) {
        for (int l = 0; l < 3; ++l) {
            const unsigned short* Ax =
                (l == 0) ? (Xb + (long long)t * 1024)
                         : (hbuf + (((l - 1) * 2 + ((t + 1) & 1)) * 65536));
            long long axs = (l == 0) ? (long long)256 * 1024 : 1024;
            lstm_step_kernel<<<16, 256, 0, stream>>>(
                Ax, axs,
                Wihb + (long long)l * 4096 * 1024,
                Whhb + (long long)l * 4096 * 1024,
                bias + l * 4096,
                hbuf + ((l * 2 + (t & 1)) * 65536),        // h read (t-1)
                cT_base + l * 65536,                       // c state in-place
                hbuf + ((l * 2 + ((t + 1) & 1)) * 65536),  // h write (t)
                hT_base + l * 65536,
                (l == 2) ? (pre_out + (long long)t * 1024) : (float*)nullptr,
                (long long)256 * 1024);
        }
    }
}
